// Net_2070174236749
// MI455X (gfx1250) — matrix operand order repaired and checked by reading, not compiled, not run on hardware
//
#include <hip/hip_runtime.h>
#include <hip/hip_bf16.h>
#include <math.h>

// ---------------------------------------------------------------------------
// Sizes (match reference)
#define BATCH 32
#define NMAX_ 512
#define NNODE 8192
#define FDIM 128
#define HDIM 512
#define LLAY 3
#define GGRP 8
#define NEDGE 131072
#define MROWS (BATCH * NMAX_)   // 16384
// GEMM register blocking: 32x64 output per wave
#define MT 2
#define NT 4
// ---------------------------------------------------------------------------

typedef __attribute__((ext_vector_type(8)))  _Float16 v8h;
typedef __attribute__((ext_vector_type(16))) _Float16 v16h;
typedef __attribute__((ext_vector_type(8)))  float    v8f;

union U16 { v16h v; v8h h[2]; };

__device__ __forceinline__ float dsigmoid(float x) {
    return 1.f / (1.f + __expf(-x));
}

// ---------------------------------------------------------------------------
// Graph densification kernels
// ---------------------------------------------------------------------------
__global__ void k_boundaries(const int* __restrict__ batch, int* __restrict__ starts, int n) {
    int i = blockIdx.x * 256 + threadIdx.x;
    if (i >= n) return;
    if (i == 0 || batch[i] != batch[i - 1]) starts[batch[i]] = i;
}

// one block (128 threads) per node: scatter x row -> dense f16, set mask/pos
__global__ __launch_bounds__(128) void k_scatter(const float* __restrict__ x,
                                                 const int* __restrict__ batch,
                                                 const int* __restrict__ starts,
                                                 _Float16* __restrict__ x16,
                                                 float* __restrict__ mask,
                                                 int* __restrict__ pos) {
    int i = blockIdx.x;                 // node
    int t = threadIdx.x;                // feature (F = 128)
    int b = batch[i];
    int p = i - starts[b];
    int row = b * NMAX_ + p;
    x16[(long long)row * FDIM + t] = (_Float16)x[(long long)i * FDIM + t];
    if (t == 0) { mask[row] = 1.f; pos[i] = p; }
}

__global__ void k_edges(const int* __restrict__ edge, const int* __restrict__ batch,
                        const int* __restrict__ pos, float* __restrict__ A) {
    int e = blockIdx.x * 256 + threadIdx.x;
    if (e >= NEDGE) return;
    int s = edge[e];
    int d = edge[NEDGE + e];
    int g = batch[s];
    A[((long long)(g * NMAX_ + pos[s])) * NMAX_ + pos[d]] = 1.f;
}

__global__ void k_selfloop(const float* __restrict__ mask, float* __restrict__ A) {
    int r = blockIdx.x * 256 + threadIdx.x;   // 0..MROWS-1
    if (r >= MROWS) return;
    if (mask[r] != 0.f) A[(long long)r * NMAX_ + (r & (NMAX_ - 1))] = 1.f;
}

// degree = row sum of A; one block (128 thr) per row
__global__ __launch_bounds__(128) void k_rowdeg(const float* __restrict__ A, float* __restrict__ deg) {
    __shared__ float sh[128];
    int row = blockIdx.x, t = threadIdx.x;
    const float* Ar = A + (long long)row * NMAX_;
    float s = 0.f;
    for (int j = t; j < NMAX_; j += 128) s += Ar[j];
    sh[t] = s; __syncthreads();
    for (int st = 64; st > 0; st >>= 1) { if (t < st) sh[t] += sh[t + st]; __syncthreads(); }
    if (t == 0) deg[row] = sh[0];
}

// W [K,N] f32 -> WT [N,K] f16
__global__ void k_transpose_f16(const float* __restrict__ W, _Float16* __restrict__ WT, int K, int N) {
    int tid = blockIdx.x * 256 + threadIdx.x;
    if (tid >= K * N) return;
    int k = tid / N, n = tid - k * N;
    WT[(long long)n * K + k] = (_Float16)W[tid];
}

// hg = h * sigmoid(deg*Wd + bd); write row-major f16 and per-graph transposed f16
__global__ void k_gate(const float* __restrict__ h, const float* __restrict__ deg,
                       const float* __restrict__ Wd, const float* __restrict__ bd,
                       _Float16* __restrict__ hg16, _Float16* __restrict__ hg16T) {
    long long tid = (long long)blockIdx.x * 256 + threadIdx.x;   // MROWS*HDIM
    int r = (int)(tid >> 9);
    int c = (int)(tid & 511);
    float g = dsigmoid(deg[r] * Wd[c] + bd[c]);
    float v = h[tid] * g;
    hg16[tid] = (_Float16)v;
    int b = r >> 9, n = r & 511;
    hg16T[(((long long)b * HDIM + c) << 9) + n] = (_Float16)v;
}

// ---------------------------------------------------------------------------
// WMMA NT GEMM, 32x64 register-blocked per wave:
//   C[b] = act(alpha * A16[b](MxK) * BT16[b](NxK)^T  [*mask] [+bias] [+Cprev])
// Per k-step: 2 A fragments + 4 B fragments -> 8 x v_wmma_f32_16x16x32_f16
// ---------------------------------------------------------------------------
__global__ __launch_bounds__(32) void k_wmma_gemm_nt(
    const _Float16* __restrict__ A, long long sA,
    const _Float16* __restrict__ BT, long long sB,
    float* __restrict__ Cf, _Float16* __restrict__ Ch, long long sC,
    const float* __restrict__ bias,
    const float* __restrict__ mmask,
    float alpha, int accum, int act,
    int M, int Nc, int K) {
    int lane = threadIdx.x;
    int n0 = blockIdx.x * (16 * NT);
    int m0 = blockIdx.y * (16 * MT);
    long long b = blockIdx.z;
    const _Float16* Ab = A + b * sA;
    const _Float16* Bb = BT + b * sB;

    v8f acc[MT][NT];
#pragma unroll
    for (int mt = 0; mt < MT; ++mt)
#pragma unroll
        for (int nt = 0; nt < NT; ++nt)
#pragma unroll
            for (int i = 0; i < 8; ++i) acc[mt][nt][i] = 0.f;

    int lm = lane & 15;        // row (A) / col (B) inside 16-tile
    int hi = lane >> 4;        // lane-half selector

    for (int k0 = 0; k0 < K; k0 += 32) {
        U16 ua[MT], ub[NT];
#pragma unroll
        for (int mt = 0; mt < MT; ++mt) {
            // A fragment: 16 contiguous K per half
            const _Float16* pa = Ab + (long long)(m0 + mt * 16 + lm) * K + k0 + 16 * hi;
            ua[mt].h[0] = *(const v8h*)(pa);
            ua[mt].h[1] = *(const v8h*)(pa + 8);
        }
#pragma unroll
        for (int nt = 0; nt < NT; ++nt) {
            // B fragment from BT (row n = column of B): 16 contiguous K per half
            const _Float16* pb = Bb + (long long)(n0 + nt * 16 + lm) * K + k0 + 16 * hi;
            ub[nt].h[0] = *(const v8h*)(pb);
            ub[nt].h[1] = *(const v8h*)(pb + 8);
        }
#pragma unroll
        for (int mt = 0; mt < MT; ++mt)
#pragma unroll
            for (int nt = 0; nt < NT; ++nt)
                acc[mt][nt] = __builtin_amdgcn_wmma_f32_16x16x32_f16(
                    false, ua[mt].v, false, ub[nt].v, (short)0, acc[mt][nt],
                    false, false);
    }

    long long cb = b * sC;
#pragma unroll
    for (int mt = 0; mt < MT; ++mt) {
#pragma unroll
        for (int nt = 0; nt < NT; ++nt) {
#pragma unroll
            for (int v = 0; v < 8; ++v) {
                int row = m0 + mt * 16 + v + 8 * hi;   // C layout: VGPR v -> M=v (+8 upper)
                int col = n0 + nt * 16 + lm;
                long long idx = cb + (long long)row * Nc + col;
                float xv = acc[mt][nt][v] * alpha;
                if (bias)  xv += bias[col];
                if (mmask) xv *= mmask[idx];
                if (accum) xv += Cf[idx];
                if (act == 1) xv = dsigmoid(xv);
                if (Cf) Cf[idx] = xv;
                if (Ch) Ch[idx] = (_Float16)xv;
            }
        }
    }
}

// attn = scores / (rowsum + 1e-6), f16 out; one block (128 thr) per row
__global__ __launch_bounds__(128) void k_rownorm(const float* __restrict__ S, _Float16* __restrict__ attn) {
    __shared__ float sh[128];
    int row = blockIdx.x, t = threadIdx.x;
    const float* Sr = S + (long long)row * NMAX_;
    float s = 0.f;
    for (int j = t; j < NMAX_; j += 128) s += Sr[j];
    sh[t] = s; __syncthreads();
    for (int st = 64; st > 0; st >>= 1) { if (t < st) sh[t] += sh[t + st]; __syncthreads(); }
    float inv = 1.f / (sh[0] + 1e-6f);
    __syncthreads();
    _Float16* Ar = attn + (long long)row * NMAX_;
    for (int j = t; j < NMAX_; j += 128) Ar[j] = (_Float16)(Sr[j] * inv);
}

// ---------------------------------------------------------------------------
// Fused epilogue: L2-normalize -> relu -> h ; LN -> grouped FFN -> h += gamma*ffn
// block = 256 threads, 8 rows per block
// ---------------------------------------------------------------------------
__device__ __forceinline__ float block_reduce_256(float v, float* sh) {
    int t = threadIdx.x;
    sh[t] = v; __syncthreads();
#pragma unroll
    for (int s = 128; s > 0; s >>= 1) { if (t < s) sh[t] += sh[t + s]; __syncthreads(); }
    float r = sh[0]; __syncthreads();
    return r;
}

__global__ __launch_bounds__(256) void k_postffn(
    const float* __restrict__ tmp, float* __restrict__ h,
    const float* __restrict__ ln_g, const float* __restrict__ ln_b,
    const float* __restrict__ up_w, const float* __restrict__ up_b,
    const float* __restrict__ up_ln_g, const float* __restrict__ up_ln_b,
    const float* __restrict__ dn_w, const float* __restrict__ dn_b,
    const float* __restrict__ dn_ln_g, const float* __restrict__ dn_ln_b,
    const float* __restrict__ gamma) {
    __shared__ float sh[256];
    __shared__ float hn[8][HDIM];
    __shared__ float yy[8][256];
    __shared__ float zz[8][64];
    __shared__ float mu[8], rs[8];
    int t = threadIdx.x;
    long long row0 = (long long)blockIdx.x * 8;

    // L2 normalize + relu -> h ; LayerNorm -> hn (LDS)
    for (int r = 0; r < 8; ++r) {
        long long row = row0 + r;
        float v0 = tmp[row * HDIM + t];
        float v1 = tmp[row * HDIM + t + 256];
        float ss = block_reduce_256(v0 * v0 + v1 * v1, sh);
        float scale = 1.f / fmaxf(sqrtf(ss), 1e-12f);
        float h0 = fmaxf(v0 * scale, 0.f);
        float h1 = fmaxf(v1 * scale, 0.f);
        h[row * HDIM + t]       = h0;
        h[row * HDIM + t + 256] = h1;
        float mean = block_reduce_256(h0 + h1, sh) * (1.f / 512.f);
        float d0 = h0 - mean, d1 = h1 - mean;
        float var = block_reduce_256(d0 * d0 + d1 * d1, sh) * (1.f / 512.f);
        float rstd = rsqrtf(var + 1e-5f);
        hn[r][t]       = d0 * rstd * ln_g[t] + ln_b[t];
        hn[r][t + 256] = d1 * rstd * ln_g[t + 256] + ln_b[t + 256];
    }
    __syncthreads();

    for (int g = 0; g < GGRP; ++g) {
        // up: thread t -> output column o=t (256 outs), 8 rows
        float acc[8];
        float bb = up_b[g * 256 + t];
#pragma unroll
        for (int r = 0; r < 8; ++r) acc[r] = bb;
        const float* wcol = up_w + (long long)g * 64 * 256 + t;   // up_w[g][i][t]
        for (int i = 0; i < 64; ++i) {
            float w = wcol[(long long)i * 256];
#pragma unroll
            for (int r = 0; r < 8; ++r) acc[r] += hn[r][g * 64 + i] * w;
        }
#pragma unroll
        for (int r = 0; r < 8; ++r) yy[r][t] = acc[r];
        __syncthreads();
        if (t < 8) {
            float s = 0.f, s2 = 0.f;
            for (int o = 0; o < 256; ++o) { float v = yy[t][o]; s += v; s2 += v * v; }
            float m = s * (1.f / 256.f);
            mu[t] = m;
            rs[t] = rsqrtf(fmaxf(s2 * (1.f / 256.f) - m * m, 0.f) + 1e-5f);
        }
        __syncthreads();
        {
            float lg = up_ln_g[t], lb = up_ln_b[t];
#pragma unroll
            for (int r = 0; r < 8; ++r)
                yy[r][t] = fmaxf((yy[r][t] - mu[r]) * rs[r] * lg + lb, 0.f);
        }
        __syncthreads();
        // down: thread -> (rr = t/64, o2 = t%64); covers rows rr and rr+4
        int rr = t >> 6, o2 = t & 63;
        float a0 = dn_b[g * 64 + o2];
        float a1 = a0;
        const float* dw = dn_w + (long long)g * 256 * 64 + o2;    // dn_w[g][o][o2]
        for (int o = 0; o < 256; ++o) {
            float w = dw[(long long)o * 64];
            a0 += yy[rr][o] * w;
            a1 += yy[rr + 4][o] * w;
        }
        zz[rr][o2]     = a0;
        zz[rr + 4][o2] = a1;
        __syncthreads();
        if (t < 8) {
            float s = 0.f, s2 = 0.f;
            for (int o = 0; o < 64; ++o) { float v = zz[t][o]; s += v; s2 += v * v; }
            float m = s * (1.f / 64.f);
            mu[t] = m;
            rs[t] = rsqrtf(fmaxf(s2 * (1.f / 64.f) - m * m, 0.f) + 1e-5f);
        }
        __syncthreads();
        {
            int r2 = t >> 5, oa = t & 31;      // 8 rows x 32 threads, 2 outputs each
            long long rowg = row0 + r2;
#pragma unroll
            for (int q = 0; q < 2; ++q) {
                int ob = oa + q * 32;
                float val = (zz[r2][ob] - mu[r2]) * rs[r2] * dn_ln_g[ob] + dn_ln_b[ob];
                int c = g * 64 + ob;
                h[rowg * HDIM + c] += gamma[c] * val;
            }
        }
        __syncthreads();
    }
}

// masked sum pool: out[b][c] = sum_n mask * h
__global__ void k_pool(const float* __restrict__ h, const float* __restrict__ mask,
                       float* __restrict__ out) {
    int tid = blockIdx.x * 256 + threadIdx.x;   // B*H
    int b = tid >> 9, c = tid & 511;
    const float* hb = h + ((long long)b * NMAX_) * HDIM + c;
    const float* mb = mask + b * NMAX_;
    float s = 0.f;
    for (int n = 0; n < NMAX_; ++n)
        if (mb[n] != 0.f) s += hb[(long long)n * HDIM];
    out[tid] = s;
}

// ---------------------------------------------------------------------------
extern "C" void kernel_launch(void* const* d_in, const int* in_sizes, int n_in,
                              void* d_out, int out_size, void* d_ws, size_t ws_size,
                              hipStream_t stream) {
    (void)in_sizes; (void)n_in; (void)out_size; (void)ws_size;

    const float* x       = (const float*)d_in[0];
    const int*   edge    = (const int*)d_in[1];
    const int*   batch   = (const int*)d_in[2];
    const float* W_in    = (const float*)d_in[3];
    const float* b_in    = (const float*)d_in[4];
    const float* Wqk     = (const float*)d_in[5];
    const float* bqk     = (const float*)d_in[6];
    const float* Wd      = (const float*)d_in[7];
    const float* bd      = (const float*)d_in[8];
    const float* Wl      = (const float*)d_in[9];
    const float* bl      = (const float*)d_in[10];
    const float* Wr      = (const float*)d_in[11];
    const float* gamma   = (const float*)d_in[12];
    const float* ln_g    = (const float*)d_in[13];
    const float* ln_b    = (const float*)d_in[14];
    const float* up_w    = (const float*)d_in[15];
    const float* up_b    = (const float*)d_in[16];
    const float* up_ln_g = (const float*)d_in[17];
    const float* up_ln_b = (const float*)d_in[18];
    const float* dn_w    = (const float*)d_in[19];
    const float* dn_b    = (const float*)d_in[20];
    const float* dn_ln_g = (const float*)d_in[21];
    const float* dn_ln_b = (const float*)d_in[22];

    // workspace carve (256B aligned)
    size_t off = 0;
    char* base = (char*)d_ws;
    auto carve = [&](size_t bytes) -> char* {
        off = (off + 255) & ~(size_t)255;
        char* p = base + off;
        off += bytes;
        return p;
    };
    int*       starts  = (int*)carve(BATCH * sizeof(int));
    int*       pos     = (int*)carve(NNODE * sizeof(int));
    float*     mask    = (float*)carve((size_t)MROWS * sizeof(float));
    _Float16*  x16     = (_Float16*)carve((size_t)MROWS * FDIM * sizeof(_Float16));
    float*     Aadj    = (float*)carve((size_t)MROWS * NMAX_ * sizeof(float));
    float*     deg     = (float*)carve((size_t)MROWS * sizeof(float));
    float*     hbuf    = (float*)carve((size_t)MROWS * HDIM * sizeof(float));
    _Float16*  hg16    = (_Float16*)carve((size_t)MROWS * HDIM * sizeof(_Float16));
    _Float16*  hg16T   = (_Float16*)carve((size_t)MROWS * HDIM * sizeof(_Float16));
    _Float16*  QK16    = (_Float16*)carve((size_t)MROWS * HDIM * sizeof(_Float16));
    float*     scores  = (float*)carve((size_t)MROWS * NMAX_ * sizeof(float));
    _Float16*  attn16  = (_Float16*)carve((size_t)MROWS * NMAX_ * sizeof(_Float16));
    _Float16*  ao16    = (_Float16*)carve((size_t)MROWS * HDIM * sizeof(_Float16));
    float*     tmpb    = (float*)carve((size_t)MROWS * HDIM * sizeof(float));
    _Float16*  WinT    = (_Float16*)carve((size_t)HDIM * FDIM * sizeof(_Float16));
    _Float16*  WqkT    = (_Float16*)carve((size_t)LLAY * HDIM * HDIM * sizeof(_Float16));
    _Float16*  WlT     = (_Float16*)carve((size_t)LLAY * HDIM * HDIM * sizeof(_Float16));
    _Float16*  WrT     = (_Float16*)carve((size_t)LLAY * HDIM * HDIM * sizeof(_Float16));

    // zero-init buffers that are sparsely written
    hipMemsetAsync(Aadj, 0, (size_t)MROWS * NMAX_ * sizeof(float), stream);
    hipMemsetAsync(x16, 0, (size_t)MROWS * FDIM * sizeof(_Float16), stream);
    hipMemsetAsync(mask, 0, (size_t)MROWS * sizeof(float), stream);

    // densify
    k_boundaries<<<(NNODE + 255) / 256, 256, 0, stream>>>(batch, starts, NNODE);
    k_scatter<<<NNODE, 128, 0, stream>>>(x, batch, starts, x16, mask, pos);
    k_edges<<<NEDGE / 256, 256, 0, stream>>>(edge, batch, pos, Aadj);
    k_selfloop<<<MROWS / 256, 256, 0, stream>>>(mask, Aadj);
    k_rowdeg<<<MROWS, 128, 0, stream>>>(Aadj, deg);

    // weight conversion (transposed f16)
    k_transpose_f16<<<(FDIM * HDIM + 255) / 256, 256, 0, stream>>>(W_in, WinT, FDIM, HDIM);
    for (int l = 0; l < LLAY; ++l) {
        long long wo = (long long)l * HDIM * HDIM;
        k_transpose_f16<<<(HDIM * HDIM + 255) / 256, 256, 0, stream>>>(Wqk + wo, WqkT + wo, HDIM, HDIM);
        k_transpose_f16<<<(HDIM * HDIM + 255) / 256, 256, 0, stream>>>(Wl + wo, WlT + wo, HDIM, HDIM);
        k_transpose_f16<<<(HDIM * HDIM + 255) / 256, 256, 0, stream>>>(Wr + wo, WrT + wo, HDIM, HDIM);
    }

    const float inv_sqrt_h = 0.044194173824159216f;   // 1/sqrt(512)
    dim3 gShared(HDIM / (16 * NT), MROWS / (16 * MT), 1);       // 8 x 512
    dim3 gBatched(NMAX_ / (16 * NT), NMAX_ / (16 * MT), BATCH); // 8 x 16 x 32

    // input projection: h = x_dense @ W_in + b_in
    k_wmma_gemm_nt<<<gShared, 32, 0, stream>>>(
        x16, 0, WinT, 0, hbuf, nullptr, 0, b_in, nullptr,
        1.f, 0, 0, MROWS, HDIM, FDIM);

    for (int l = 0; l < LLAY; ++l) {
        long long wo = (long long)l * HDIM * HDIM;
        // degree gate -> hg (f16 row-major + per-graph transposed)
        k_gate<<<(MROWS * HDIM) / 256, 256, 0, stream>>>(
            hbuf, deg, Wd + l * HDIM, bd + l * HDIM, hg16, hg16T);
        // QK = sigmoid(hg @ Wqk + bqk)  (f16 out)
        k_wmma_gemm_nt<<<gShared, 32, 0, stream>>>(
            hg16, 0, WqkT + wo, 0, nullptr, QK16, 0, bqk + l * HDIM, nullptr,
            1.f, 0, 1, MROWS, HDIM, HDIM);
        // scores[b] = (QK QK^T) * inv_sqrt_h * A   (batched, mask fused)
        k_wmma_gemm_nt<<<gBatched, 32, 0, stream>>>(
            QK16, (long long)NMAX_ * NMAX_, QK16, (long long)NMAX_ * NMAX_,
            scores, nullptr, (long long)NMAX_ * NMAX_, nullptr, Aadj,
            inv_sqrt_h, 0, 0, NMAX_, NMAX_, HDIM);
        // attn = scores / (rowsum + 1e-6)  (f16)
        k_rownorm<<<MROWS, 128, 0, stream>>>(scores, attn16);
        // attn_out[b] = attn @ hg   (f16 out)
        k_wmma_gemm_nt<<<gBatched, 32, 0, stream>>>(
            attn16, (long long)NMAX_ * NMAX_, hg16T, (long long)HDIM * NMAX_,
            nullptr, ao16, (long long)NMAX_ * HDIM, nullptr, nullptr,
            1.f, 0, 0, NMAX_, HDIM, NMAX_);
        // tmp = attn_out @ Wl + bl
        k_wmma_gemm_nt<<<gShared, 32, 0, stream>>>(
            ao16, 0, WlT + wo, 0, tmpb, nullptr, 0, bl + l * HDIM, nullptr,
            1.f, 0, 0, MROWS, HDIM, HDIM);
        // tmp += hg @ Wr
        k_wmma_gemm_nt<<<gShared, 32, 0, stream>>>(
            hg16, 0, WrT + wo, 0, tmpb, nullptr, 0, nullptr, nullptr,
            1.f, 1, 0, MROWS, HDIM, HDIM);
        // L2norm/relu -> h ; LN ; grouped FFN ; h += gamma * ffn
        k_postffn<<<MROWS / 8, 256, 0, stream>>>(
            tmpb, hbuf,
            ln_g + l * HDIM, ln_b + l * HDIM,
            up_w + (long long)l * GGRP * 64 * 256, up_b + (long long)l * GGRP * 256,
            up_ln_g + l * 256, up_ln_b + l * 256,
            dn_w + (long long)l * GGRP * 256 * 64, dn_b + (long long)l * GGRP * 64,
            dn_ln_g + l * 64, dn_ln_b + l * 64,
            gamma + l * HDIM);
    }

    // masked sum pool -> [B, H]
    k_pool<<<(BATCH * HDIM) / 256, 256, 0, stream>>>(hbuf, mask, (float*)d_out);
}
